// GRA_Enhanced_35673998361168
// MI455X (gfx1250) — compile-verified, matrix operand order checked by reading, not run on hardware
//
#include <hip/hip_runtime.h>
#include <hip/hip_bf16.h>

// ---------------------------------------------------------------------------
// CDNA5 (gfx1250) transformer forward.
// GEMMs use V_WMMA_F32_16X16X32_BF16 (wave32), fp32->bf16 conversion during
// LDS staging, fp32 accumulation. LDS tiles are stored pre-swizzled into the
// WMMA per-lane fragment layout (ISA 7.12.2) so fragment reads are contiguous
// 32B ds_load_b128 pairs. 128x64 block tile, 8 waves, 32x32 per wave
// (4 WMMA / wave / K-step), double-buffered LDS with register prefetch.
// ---------------------------------------------------------------------------

typedef __attribute__((ext_vector_type(16))) __bf16 v16bf;
typedef __attribute__((ext_vector_type(8)))  __bf16 v8bf;
typedef __attribute__((ext_vector_type(8)))  float  v8f;

#define BM 128
#define BN 64
#define BKT 32

// Fragment-layout helpers (16-bit operand, 16x32 tile, ISA 7.12.2):
// element e of lane l holds (row = l%16, k = (e/8)*16 + (l/16)*8 + e%8).
// Inverse: given k in [0,32): hi=k>=16, w=k&15 -> lane_half=w>>3, e=(hi<<3)|(w&7).

__global__ void k_gemm(const float* __restrict__ A, const float* __restrict__ B,
                       float* __restrict__ C,
                       int M, int N, int K,
                       long long sAm, long long sAk,
                       long long sBk, long long sBn,
                       long long sCm, long long sCn,
                       int Hn,
                       long long sAb, long long sAh,
                       long long sBb, long long sBh,
                       long long sCb, long long sCh,
                       float alpha, int epi)
{
    // [buf][subtile][lane][elem] ; A: 8 m-subtiles, B: 4 n-subtiles
    __shared__ alignas(32) __bf16 Asf[2][BM / 16][32][16];
    __shared__ alignas(32) __bf16 Bsf[2][BN / 16][32][16];

    const int z  = blockIdx.z;
    const int zb = z / Hn, zh = z % Hn;
    const float* Ab = A + zb * sAb + zh * sAh;
    const float* Bb = B + zb * sBb + zh * sBh;
    float*       Cb = C + zb * sCb + zh * sCh;

    const int m0 = blockIdx.x * BM;
    const int n0 = blockIdx.y * BN;

    const int tid  = threadIdx.x;
    const int lane = tid & 31, wave = tid >> 5;
    const int wm2 = wave >> 1, wn2 = wave & 1;      // wave -> 32x32 macro tile
    const int lh = lane >> 4, lm = lane & 15;

    // A staging: thread -> (row = tid/2 in [0,128), 16 consecutive k)
    const int arow = tid >> 1, ak0 = (tid & 1) * 16;
    const int amt = arow >> 4, ar = arow & 15;
    // B staging: thread -> (k = tid/8 in [0,32), 8 consecutive n)
    const int bkr = tid >> 3, bn0 = (tid & 7) * 8;
    const int bhi = (bkr >= 16) ? 1 : 0, bw = bkr & 15;
    const int blh = bw >> 3, be = (bhi << 3) | (bw & 7);
    const int bnt = bn0 >> 4;

    float areg[16];
    float breg[8];

    auto load_tiles = [&](int k0) {
        const float* ap = Ab + (long long)(m0 + arow) * sAm + (long long)(k0 + ak0) * sAk;
        if (sAk == 1) {
            const float4* p4 = (const float4*)ap;
#pragma unroll
            for (int i = 0; i < 4; ++i) {
                float4 v = p4[i];
                areg[4 * i + 0] = v.x; areg[4 * i + 1] = v.y;
                areg[4 * i + 2] = v.z; areg[4 * i + 3] = v.w;
            }
        } else {
#pragma unroll
            for (int i = 0; i < 16; ++i) areg[i] = ap[(long long)i * sAk];
        }
        const float* bp = Bb + (long long)(k0 + bkr) * sBk + (long long)(n0 + bn0) * sBn;
        if (sBn == 1) {
            const float4* p4 = (const float4*)bp;
#pragma unroll
            for (int j = 0; j < 2; ++j) {
                float4 v = p4[j];
                breg[4 * j + 0] = v.x; breg[4 * j + 1] = v.y;
                breg[4 * j + 2] = v.z; breg[4 * j + 3] = v.w;
            }
        } else {
#pragma unroll
            for (int j = 0; j < 8; ++j) breg[j] = bp[(long long)j * sBn];
        }
    };

    auto store_tiles = [&](int buf) {
        // A: two contiguous groups of 8 bf16 -> 16B vector stores
#pragma unroll
        for (int g = 0; g < 2; ++g) {
            const int ks = ak0 + g * 8;              // 0,8,16,24
            const int hi = (ks >= 16) ? 1 : 0;
            const int w = ks & 15;
            const int laneA = ((w >> 3) << 4) | ar;
            const int e0 = hi << 3;
            v8bf pk;
#pragma unroll
            for (int i = 0; i < 8; ++i) pk[i] = (__bf16)areg[g * 8 + i];
            *(v8bf*)&Asf[buf][amt][laneA][e0] = pk;
        }
        // B: 8 scalar bf16 scatter stores (lane varies with n)
#pragma unroll
        for (int j = 0; j < 8; ++j) {
            const int c = (bn0 & 15) + j;
            Bsf[buf][bnt][(blh << 4) | c][be] = (__bf16)breg[j];
        }
    };

    v8f acc[2][2];
    acc[0][0] = (v8f){}; acc[0][1] = (v8f){};
    acc[1][0] = (v8f){}; acc[1][1] = (v8f){};

    const int nk = K / BKT;
    load_tiles(0);
    store_tiles(0);
    __syncthreads();

    for (int ki = 0; ki < nk; ++ki) {
        const int buf = ki & 1;
        if (ki + 1 < nk) load_tiles((ki + 1) * BKT);

        v16bf a0 = *(const v16bf*)&Asf[buf][wm2 * 2 + 0][lane][0];
        v16bf a1 = *(const v16bf*)&Asf[buf][wm2 * 2 + 1][lane][0];
        v16bf b0 = *(const v16bf*)&Bsf[buf][wn2 * 2 + 0][lane][0];
        v16bf b1 = *(const v16bf*)&Bsf[buf][wn2 * 2 + 1][lane][0];

        acc[0][0] = __builtin_amdgcn_wmma_f32_16x16x32_bf16(false, a0, false, b0,
                        (short)0, acc[0][0], false, false);
        acc[0][1] = __builtin_amdgcn_wmma_f32_16x16x32_bf16(false, a0, false, b1,
                        (short)0, acc[0][1], false, false);
        acc[1][0] = __builtin_amdgcn_wmma_f32_16x16x32_bf16(false, a1, false, b0,
                        (short)0, acc[1][0], false, false);
        acc[1][1] = __builtin_amdgcn_wmma_f32_16x16x32_bf16(false, a1, false, b1,
                        (short)0, acc[1][1], false, false);

        if (ki + 1 < nk) store_tiles(buf ^ 1);
        __syncthreads();
    }

#pragma unroll
    for (int i = 0; i < 2; ++i)
#pragma unroll
        for (int j = 0; j < 2; ++j)
#pragma unroll
            for (int r = 0; r < 8; ++r) {
                const int m = m0 + (wm2 * 2 + i) * 16 + lh * 8 + r;
                const int n = n0 + (wn2 * 2 + j) * 16 + lm;
                float v = acc[i][j][r] * alpha;
                if (epi == 1)  // exact GELU, matches jax.nn.gelu(approximate=False)
                    v = 0.5f * v * (1.0f + erff(v * 0.70710678118654752f));
                Cb[(long long)m * sCm + (long long)n * sCn] = v;
            }
}

// Row softmax with padding + optional causal mask.
// grid = (Sq, 1, B*Hn); one 256-thread block per score row. Sk <= 1024.
__global__ void k_softmax(float* __restrict__ sc, const int* __restrict__ key_ids,
                          int Sq, int Sk, int Hn, int causal)
{
    __shared__ float sh[256];
    const int q = blockIdx.x;
    const int z = blockIdx.z;
    const int b = z / Hn;
    float* row = sc + ((long long)z * Sq + q) * (long long)Sk;
    const int t = threadIdx.x;

    float vreg[4];
    float mx = -3.0e38f;
    int idx = 0;
    for (int k = t; k < Sk; k += 256, ++idx) {
        float v = row[k];
        const bool masked = (key_ids[(long long)b * Sk + k] == 0) ||
                            (causal && (k > q));
        v = masked ? -1.0e9f : v;        // reference: where(mask, -1e9, scores)
        vreg[idx] = v;
        mx = fmaxf(mx, v);
    }
    sh[t] = mx; __syncthreads();
    for (int s = 128; s; s >>= 1) { if (t < s) sh[t] = fmaxf(sh[t], sh[t + s]); __syncthreads(); }
    mx = sh[0]; __syncthreads();

    float sum = 0.f;
    idx = 0;
    for (int k = t; k < Sk; k += 256, ++idx) {
        const float e = expf(vreg[idx] - mx);
        vreg[idx] = e;
        sum += e;
    }
    sh[t] = sum; __syncthreads();
    for (int s = 128; s; s >>= 1) { if (t < s) sh[t] += sh[t + s]; __syncthreads(); }
    const float inv = 1.0f / sh[0];
    __syncthreads();

    idx = 0;
    for (int k = t; k < Sk; k += 256, ++idx)
        row[k] = vreg[idx] * inv;
}

// out = LayerNorm(a + res) * gamma + beta ; gamma_beta = [gamma(D), beta(D)].
// One 256-thread block per row. Safe when out aliases res.
__global__ void k_add_ln(const float* __restrict__ a, const float* __restrict__ res,
                         const float* __restrict__ gamma_beta,
                         float* __restrict__ out, int D)
{
    __shared__ float sh[256];
    const long long row = blockIdx.x;
    const float* ap = a   + row * D;
    const float* rp = res + row * D;
    float*       op = out + row * D;
    const int t = threadIdx.x;

    float s = 0.f;
    for (int i = t; i < D; i += 256) s += ap[i] + rp[i];
    sh[t] = s; __syncthreads();
    for (int k = 128; k; k >>= 1) { if (t < k) sh[t] += sh[t + k]; __syncthreads(); }
    const float mean = sh[0] / (float)D; __syncthreads();

    float v = 0.f;
    for (int i = t; i < D; i += 256) { const float d = ap[i] + rp[i] - mean; v += d * d; }
    sh[t] = v; __syncthreads();
    for (int k = 128; k; k >>= 1) { if (t < k) sh[t] += sh[t + k]; __syncthreads(); }
    const float rstd = rsqrtf(sh[0] / (float)D + 1e-5f);
    __syncthreads();

    for (int i = t; i < D; i += 256)
        op[i] = (ap[i] + rp[i] - mean) * rstd * gamma_beta[i] + gamma_beta[D + i];
}

// ---------------------------------------------------------------------------
// Host orchestration
// ---------------------------------------------------------------------------

static inline void launch_gemm(hipStream_t st, const float* A, const float* B, float* C,
                               int M, int N, int K,
                               long long sAm, long long sAk,
                               long long sBk, long long sBn,
                               long long sCm, long long sCn,
                               int nz, int Hn,
                               long long sAb, long long sAh,
                               long long sBb, long long sBh,
                               long long sCb, long long sCh,
                               float alpha, int epi)
{
    dim3 g(M / BM, N / BN, nz), b(256);
    k_gemm<<<g, b, 0, st>>>(A, B, C, M, N, K, sAm, sAk, sBk, sBn, sCm, sCn,
                            Hn, sAb, sAh, sBb, sBh, sCb, sCh, alpha, epi);
}

namespace {

constexpr int Bsz = 8, S = 512, Dm = 768, Hh = 12, DK = 64, Lc = 6, FF = 3072, Vv = 8192;
constexpr int NT = Bsz * S;               // 4096 token rows
constexpr float SCALE = 0.125f;           // 1/sqrt(64)

struct Bufs { float *X, *Y, *Q, *K, *V, *Ctx, *T, *Hb, *Sc; };

// One multi-head attention block: overwrites resbuf in place with LN(ctx@Wo + resbuf).
void run_mha(hipStream_t st, const float* qin, const float* kvin, float* resbuf,
             const float* w4, const float* lnw, const int* key_ids, int causal,
             const Bufs& bf)
{
    const long long DD = (long long)Dm * Dm;
    // Q/K/V projections: [NT,D] @ [D,D]
    launch_gemm(st, qin,  w4 + 0 * DD, bf.Q, NT, Dm, Dm, Dm, 1, Dm, 1, Dm, 1,
                1, 1, 0, 0, 0, 0, 0, 0, 1.0f, 0);
    launch_gemm(st, kvin, w4 + 1 * DD, bf.K, NT, Dm, Dm, Dm, 1, Dm, 1, Dm, 1,
                1, 1, 0, 0, 0, 0, 0, 0, 1.0f, 0);
    launch_gemm(st, kvin, w4 + 2 * DD, bf.V, NT, Dm, Dm, Dm, 1, Dm, 1, Dm, 1,
                1, 1, 0, 0, 0, 0, 0, 0, 1.0f, 0);

    // scores[b,h] = SCALE * Q[b,:,h*64:] @ K[b,:,h*64:]^T   (S x S, K-dim 64)
    launch_gemm(st, bf.Q, bf.K, bf.Sc, S, S, DK,
                /*A*/ Dm, 1, /*B: k=d stride 1, n=key stride D*/ 1, Dm,
                /*C*/ S, 1,
                Bsz * Hh, Hh,
                (long long)S * Dm, DK,             // A batch/head strides
                (long long)S * Dm, DK,             // B batch/head strides
                (long long)Hh * S * S, (long long)S * S,
                SCALE, 0);

    {   // softmax with pad (+causal) mask
        dim3 g(S, 1, Bsz * Hh), b(256);
        k_softmax<<<g, b, 0, st>>>(bf.Sc, key_ids, S, S, Hh, causal);
    }

    // ctx[b,:,h*64:] = probs[b,h] @ V[b,:,h*64:]   (S x 64, K-dim S)
    launch_gemm(st, bf.Sc, bf.V, bf.Ctx, S, DK, S,
                /*A*/ S, 1, /*B*/ Dm, 1, /*C*/ Dm, 1,
                Bsz * Hh, Hh,
                (long long)Hh * S * S, (long long)S * S,
                (long long)S * Dm, DK,
                (long long)S * Dm, DK,
                1.0f, 0);

    // output projection + residual layernorm
    launch_gemm(st, bf.Ctx, w4 + 3 * DD, bf.T, NT, Dm, Dm, Dm, 1, Dm, 1, Dm, 1,
                1, 1, 0, 0, 0, 0, 0, 0, 1.0f, 0);
    k_add_ln<<<dim3(NT), dim3(256), 0, st>>>(bf.T, resbuf, lnw, resbuf, Dm);
}

void run_ffn(hipStream_t st, float* xbuf, const float* w1, const float* w2,
             const float* lnw, const Bufs& bf)
{
    launch_gemm(st, xbuf, w1, bf.Hb, NT, FF, Dm, Dm, 1, FF, 1, FF, 1,
                1, 1, 0, 0, 0, 0, 0, 0, 1.0f, /*gelu*/ 1);
    launch_gemm(st, bf.Hb, w2, bf.T, NT, Dm, FF, FF, 1, Dm, 1, Dm, 1,
                1, 1, 0, 0, 0, 0, 0, 0, 1.0f, 0);
    k_add_ln<<<dim3(NT), dim3(256), 0, st>>>(bf.T, xbuf, lnw, xbuf, Dm);
}

} // namespace

extern "C" void kernel_launch(void* const* d_in, const int* in_sizes, int n_in,
                              void* d_out, int out_size, void* d_ws, size_t ws_size,
                              hipStream_t stream)
{
    const int*   enc_ids    = (const int*)d_in[0];
    const int*   dec_ids    = (const int*)d_in[1];
    const float* bert       = (const float*)d_in[2];
    const float* gpt        = (const float*)d_in[3];
    const float* enc_attn_w = (const float*)d_in[4];
    const float* enc_ln     = (const float*)d_in[5];   // [L,2,2,D]
    const float* enc_w1     = (const float*)d_in[6];
    const float* enc_w2     = (const float*)d_in[7];
    const float* dec_self_w = (const float*)d_in[8];
    const float* dec_cross_w= (const float*)d_in[9];
    const float* dec_ln     = (const float*)d_in[10];  // [L,3,2,D]
    const float* dec_w1     = (const float*)d_in[11];
    const float* dec_w2     = (const float*)d_in[12];
    const float* proj_w     = (const float*)d_in[13];  // [D,V]
    float* out = (float*)d_out;
    (void)in_sizes; (void)n_in; (void)out_size; (void)ws_size;

    // Workspace arena (~239 MB of fp32 scratch).
    float* wsp = (float*)d_ws;
    size_t off = 0;
    auto take = [&](size_t n) { float* p = wsp + off; off += n; return p; };
    Bufs bf;
    bf.X   = take((size_t)NT * Dm);
    bf.Y   = take((size_t)NT * Dm);
    bf.Q   = take((size_t)NT * Dm);
    bf.K   = take((size_t)NT * Dm);
    bf.V   = take((size_t)NT * Dm);
    bf.Ctx = take((size_t)NT * Dm);
    bf.T   = take((size_t)NT * Dm);
    bf.Hb  = take((size_t)NT * FF);
    bf.Sc  = take((size_t)Bsz * Hh * S * S);

    const size_t xbytes = (size_t)NT * Dm * sizeof(float);
    hipMemcpyAsync(bf.X, bert, xbytes, hipMemcpyDeviceToDevice, stream);
    hipMemcpyAsync(bf.Y, gpt,  xbytes, hipMemcpyDeviceToDevice, stream);

    const long long DD = (long long)Dm * Dm;

    // ---- encoder ----
    for (int i = 0; i < Lc; ++i) {
        const float* w4  = enc_attn_w + (long long)i * 4 * DD;
        const float* ln0 = enc_ln + ((long long)i * 2 + 0) * 2 * Dm;
        const float* ln1 = enc_ln + ((long long)i * 2 + 1) * 2 * Dm;
        run_mha(stream, bf.X, bf.X, bf.X, w4, ln0, enc_ids, /*causal*/0, bf);
        run_ffn(stream, bf.X, enc_w1 + (long long)i * Dm * FF,
                enc_w2 + (long long)i * FF * Dm, ln1, bf);
    }
    // bf.X now holds enc_out.

    // ---- decoder ----
    for (int i = 0; i < Lc; ++i) {
        const float* ws4 = dec_self_w  + (long long)i * 4 * DD;
        const float* wc4 = dec_cross_w + (long long)i * 4 * DD;
        const float* ln0 = dec_ln + ((long long)i * 3 + 0) * 2 * Dm;
        const float* ln1 = dec_ln + ((long long)i * 3 + 1) * 2 * Dm;
        const float* ln2 = dec_ln + ((long long)i * 3 + 2) * 2 * Dm;
        run_mha(stream, bf.Y, bf.Y, bf.Y, ws4, ln0, dec_ids, /*causal*/1, bf);
        run_mha(stream, bf.Y, bf.X, bf.Y, wc4, ln1, enc_ids, /*causal*/0, bf);
        run_ffn(stream, bf.Y, dec_w1 + (long long)i * Dm * FF,
                dec_w2 + (long long)i * FF * Dm, ln2, bf);
    }

    // ---- final vocab projection: [NT,D] @ [D,V] -> d_out ----
    launch_gemm(stream, bf.Y, proj_w, out, NT, Vv, Dm,
                Dm, 1, Vv, 1, Vv, 1,
                1, 1, 0, 0, 0, 0, 0, 0, 1.0f, 0);
}
